// AdvancedGCN_31988916421038
// MI455X (gfx1250) — matrix operand order repaired and checked by reading, not compiled, not run on hardware
//
#include <hip/hip_runtime.h>

#define D_IN 128
#define HID  64
#define NCLS 10
#define NGR  256

typedef __attribute__((ext_vector_type(16))) _Float16 v16h;
typedef __attribute__((ext_vector_type(8)))  float    v8f;

// ---------------- degree / normalization ----------------
__global__ __launch_bounds__(256) void gcn_deg_init(float* dinv, int n) {
    int i = blockIdx.x * 256 + threadIdx.x;
    if (i < n) dinv[i] = 1.0f;                 // self-loop contributes 1
}
__global__ __launch_bounds__(256) void gcn_deg_edge(const int* __restrict__ dst,
                                                    float* dinv, int nE) {
    int e = blockIdx.x * 256 + threadIdx.x;
    if (e < nE) atomicAdd(&dinv[dst[e]], 1.0f);
}
__global__ __launch_bounds__(256) void gcn_dinv(float* dinv, int n) {
    int i = blockIdx.x * 256 + threadIdx.x;
    if (i < n) dinv[i] = rsqrtf(dinv[i]);      // deg >= 1 always
}

// ---------------- weight f32 -> f16, packed into WMMA B-fragment order ----------
// Packed layout: wp[(((kt*4 + nt)*32 + lane)*16 + j], where the
// element is B[kt*32 + 16*(j>>3) + 8*(lane>>4) + (j&7)][nt*16 + (lane&15)].
// Each lane's 16 halves are contiguous (32 bytes) -> one v16h load (2x b128).
__global__ __launch_bounds__(256) void gcn_wpack(const float* __restrict__ w,
                                                 _Float16* __restrict__ wp, int K) {
    int i = blockIdx.x * 256 + threadIdx.x;
    if (i < K * HID) {
        int j    = i & 15;
        int lane = (i >> 4) & 31;
        int nt   = (i >> 9) & 3;
        int kt   = i >> 11;
        int half = lane >> 4, r = lane & 15;
        int v = j >> 1, e = j & 1;
        int k = kt * 32 + ((v < 4) ? 2 * v : 16 + 2 * (v - 4)) + half * 8 + e;
        int n = nt * 16 + r;
        wp[i] = (_Float16)w[k * HID + n];
    }
}

// ---------------- WMMA GEMM: C[N,64] = (relu?)A[N,K] @ B[K,64] ----------------
// One wave per 16-row tile; 4 v8f accumulators cover all 64 output columns.
// A fragment (16x32 f16, ISA 7.12.2): lane r=row, VGPR v: K = {2v | 16+2(v-4)} + 8*half
// B fragment: pre-packed (gcn_wpack), single contiguous 32B load per lane.
// C/D: VGPR i -> M = i + 8*half, N = r
template <int RELU>
__global__ __launch_bounds__(256) void gcn_gemm_wmma(const float* __restrict__ A,
                                                     const _Float16* __restrict__ Bp,
                                                     float* __restrict__ C,
                                                     int N, int K) {
    const int lane = threadIdx.x & 31;
    const int wave = threadIdx.x >> 5;
    const int half = lane >> 4;
    const int r    = lane & 15;
    const int row0 = (blockIdx.x * 8 + wave) * 16;
    if (row0 >= N) return;                     // wave-uniform (N % 16 == 0 here)

    v8f acc[4] = {};
    const float* Arow = A + (size_t)(row0 + r) * K;

    for (int k0 = 0; k0 < K; k0 += 32) {
        v16h af;
#pragma unroll
        for (int v = 0; v < 8; ++v) {
            int kb = ((v < 4) ? 2 * v : 16 + 2 * (v - 4)) + half * 8;
            float2 p = *(const float2*)(Arow + k0 + kb);
            if (RELU) { p.x = fmaxf(p.x, 0.0f); p.y = fmaxf(p.y, 0.0f); }
            af[2 * v]     = (_Float16)p.x;
            af[2 * v + 1] = (_Float16)p.y;
        }
        const _Float16* btile = Bp + (((size_t)(k0 >> 5) * 4) * 32 + lane) * 16;
#pragma unroll
        for (int nt = 0; nt < 4; ++nt) {
            v16h bf = *(const v16h*)(btile + (size_t)nt * 32 * 16);
            acc[nt] = __builtin_amdgcn_wmma_f32_16x16x32_f16(
                false, af, false, bf, (short)0, acc[nt], false, false);
        }
    }
#pragma unroll
    for (int nt = 0; nt < 4; ++nt)
#pragma unroll
        for (int i = 0; i < 8; ++i)
            C[(size_t)(row0 + half * 8 + i) * HID + nt * 16 + r] = acc[nt][i];
}

// ---------------- self-loop + bias: agg = h*dinv^2 + b (float4 per thread) ----
__global__ __launch_bounds__(256) void gcn_self(const float* __restrict__ h,
                                                const float* __restrict__ dinv,
                                                const float* __restrict__ b,
                                                float* __restrict__ agg, int n16) {
    int i = blockIdx.x * 256 + threadIdx.x;
    if (i < n16) {
        int v = i >> 4, q = (i & 15) * 4;
        float d = dinv[v];
        float d2 = d * d;
        float4 hv = *(const float4*)(h + (size_t)v * HID + q);
        const float4 bv = *(const float4*)(b + q);
        float4 o;
        o.x = hv.x * d2 + bv.x;
        o.y = hv.y * d2 + bv.y;
        o.z = hv.z * d2 + bv.z;
        o.w = hv.w * d2 + bv.w;
        *(float4*)(agg + (size_t)v * HID + q) = o;
    }
}

// ------ edge scatter: agg[dst] += h[src] * dinv[src]*dinv[dst] (float4/thread) --
__global__ __launch_bounds__(256) void gcn_scatter(const int* __restrict__ src,
                                                   const int* __restrict__ dst,
                                                   const float* __restrict__ h,
                                                   const float* __restrict__ dinv,
                                                   float* __restrict__ agg, int nE) {
    int idx = blockIdx.x * 256 + threadIdx.x;  // nE*16 = 51.2M threads
    int e = idx >> 4, q = (idx & 15) * 4;
    if (e < nE) {
        int s = src[e], d = dst[e];
        float nm = dinv[s] * dinv[d];
        float4 hv = *(const float4*)(h + (size_t)s * HID + q);
        float* ap = agg + (size_t)d * HID + q;
        atomicAdd(ap + 0, hv.x * nm);
        atomicAdd(ap + 1, hv.y * nm);
        atomicAdd(ap + 2, hv.z * nm);
        atomicAdd(ap + 3, hv.w * nm);
    }
}

// ---------------- global mean pool + linear head ----------------
__global__ __launch_bounds__(256) void gcn_pool_init(float* pooled, float* cnt) {
    int i = blockIdx.x * 256 + threadIdx.x;
    if (i < NGR * HID) pooled[i] = 0.0f;
    if (i < NGR)       cnt[i]    = 0.0f;
}
__global__ __launch_bounds__(256) void gcn_pool_acc(const float* __restrict__ agg,
                                                    const int* __restrict__ batch,
                                                    float* pooled, float* cnt, int n16) {
    int i = blockIdx.x * 256 + threadIdx.x;
    if (i < n16) {
        int v = i >> 4, q = (i & 15) * 4;
        int g = batch[v];
        float4 hv = *(const float4*)(agg + (size_t)v * HID + q);
        float* pp = pooled + (size_t)g * HID + q;
        atomicAdd(pp + 0, hv.x);
        atomicAdd(pp + 1, hv.y);
        atomicAdd(pp + 2, hv.z);
        atomicAdd(pp + 3, hv.w);
        if (q == 0) atomicAdd(&cnt[g], 1.0f);
    }
}
__global__ __launch_bounds__(32) void gcn_final(const float* __restrict__ pooled,
                                                const float* __restrict__ cnt,
                                                const float* __restrict__ Wl,
                                                const float* __restrict__ bl,
                                                float* __restrict__ out) {
    int g = blockIdx.x, t = threadIdx.x;
    if (t < NCLS) {
        float s = 0.0f;
#pragma unroll
        for (int c = 0; c < HID; ++c) s += pooled[(size_t)g * HID + c] * Wl[c * NCLS + t];
        out[(size_t)g * NCLS + t] = s / fmaxf(cnt[g], 1.0f) + bl[t];
    }
}

// ---------------- host launch ----------------
extern "C" void kernel_launch(void* const* d_in, const int* in_sizes, int n_in,
                              void* d_out, int out_size, void* d_ws, size_t ws_size,
                              hipStream_t stream) {
    const float* x     = (const float*)d_in[0];
    const int*   ei    = (const int*)d_in[1];
    const int*   batch = (const int*)d_in[2];
    const float* W1    = (const float*)d_in[3];
    const float* b1    = (const float*)d_in[4];
    const float* W2    = (const float*)d_in[5];
    const float* b2    = (const float*)d_in[6];
    const float* W3    = (const float*)d_in[7];
    const float* b3    = (const float*)d_in[8];
    const float* Wl    = (const float*)d_in[9];
    const float* bl    = (const float*)d_in[10];
    float* out = (float*)d_out;

    const int N  = in_sizes[0] / D_IN;
    const int nE = in_sizes[1] / 2;
    const int* src = ei;
    const int* dst = ei + nE;

    char* ws = (char*)d_ws;
    size_t off = 0;
    auto take = [&](size_t bytes) -> char* {
        char* p = ws + off;
        off = (off + bytes + 255) & ~(size_t)255;
        return p;
    };
    float*    dinv   = (float*)take((size_t)N * 4);
    float*    hbuf   = (float*)take((size_t)N * HID * 4);
    float*    agg    = (float*)take((size_t)N * HID * 4);
    _Float16* wf16   = (_Float16*)take((size_t)D_IN * HID * 2);
    float*    pooled = (float*)take((size_t)NGR * HID * 4);
    float*    cnt    = (float*)take((size_t)NGR * 4);

    const int n16 = N * (HID / 4);
    auto cdiv = [](long long a, long long b) { return (unsigned)((a + b - 1) / b); };
    dim3 blk(256);
    const unsigned gGemm = cdiv((long long)N, 16 * 8);
    const unsigned gN    = cdiv(N, 256);
    const unsigned gE    = cdiv(nE, 256);
    const unsigned gN16  = cdiv(n16, 256);
    const unsigned gE16  = cdiv((long long)nE * 16, 256);

    // normalization coefficients
    gcn_deg_init<<<gN, blk, 0, stream>>>(dinv, N);
    gcn_deg_edge<<<gE, blk, 0, stream>>>(dst, dinv, nE);
    gcn_dinv<<<gN, blk, 0, stream>>>(dinv, N);

    // layer 1: x[N,128] @ W1 -> agg (bias + self-loop + edge scatter)
    gcn_wpack<<<cdiv(D_IN * HID, 256), blk, 0, stream>>>(W1, wf16, D_IN);
    gcn_gemm_wmma<0><<<gGemm, blk, 0, stream>>>(x, wf16, hbuf, N, D_IN);
    gcn_self<<<gN16, blk, 0, stream>>>(hbuf, dinv, b1, agg, n16);
    gcn_scatter<<<gE16, blk, 0, stream>>>(src, dst, hbuf, dinv, agg, nE);

    // layer 2: relu of layer-1 output fused into A-load
    gcn_wpack<<<cdiv(HID * HID, 256), blk, 0, stream>>>(W2, wf16, HID);
    gcn_gemm_wmma<1><<<gGemm, blk, 0, stream>>>(agg, wf16, hbuf, N, HID);
    gcn_self<<<gN16, blk, 0, stream>>>(hbuf, dinv, b2, agg, n16);
    gcn_scatter<<<gE16, blk, 0, stream>>>(src, dst, hbuf, dinv, agg, nE);

    // layer 3 (no relu on its output)
    gcn_wpack<<<cdiv(HID * HID, 256), blk, 0, stream>>>(W3, wf16, HID);
    gcn_gemm_wmma<1><<<gGemm, blk, 0, stream>>>(agg, wf16, hbuf, N, HID);
    gcn_self<<<gN16, blk, 0, stream>>>(hbuf, dinv, b3, agg, n16);
    gcn_scatter<<<gE16, blk, 0, stream>>>(src, dst, hbuf, dinv, agg, nE);

    // global mean pool + linear head
    gcn_pool_init<<<cdiv(NGR * HID, 256), blk, 0, stream>>>(pooled, cnt);
    gcn_pool_acc<<<gN16, blk, 0, stream>>>(agg, batch, pooled, cnt, n16);
    gcn_final<<<NGR, 32, 0, stream>>>(pooled, cnt, Wl, bl, out);
}
